// SelfAttention_48696339202635
// MI455X (gfx1250) — matrix operand order repaired and checked by reading, not compiled, not run on hardware
//
#include <hip/hip_runtime.h>

typedef __attribute__((ext_vector_type(16))) _Float16 v16h;
typedef __attribute__((ext_vector_type(8)))  _Float16 v8h;
typedef __attribute__((ext_vector_type(8)))  float    v8f;

#define NB   4
#define SEQ  2048
#define EMB  1024
#define NH   16
#define HD   64

// Workspace layout in halves (base 256B aligned by hipMalloc)
static constexpr long OFF_Q  = 0;                             // (N,H,S,D) f16
static constexpr long OFF_K  = OFF_Q  + (long)NB*NH*SEQ*HD;   // (N,H,S,D) f16
static constexpr long OFF_V  = OFF_K  + (long)NB*NH*SEQ*HD;   // (N,H,D,S) f16 (transposed)
static constexpr long OFF_O  = OFF_V  + (long)NB*NH*SEQ*HD;   // (N,S,E)  f16
static constexpr long OFF_WO = OFF_O  + (long)NB*SEQ*EMB;     // Wo^T (E,E) f16
static constexpr long OFF_WQ = OFF_WO + (long)EMB*EMB;        // Wq^T (D,D) f16
static constexpr long OFF_WK = OFF_WQ + (long)HD*HD;          // Wk^T
static constexpr long OFF_WV = OFF_WK + (long)HD*HD;          // Wv^T

__device__ __forceinline__ v8f wmma_f16(v16h a, v16h b, v8f c) {
  // (neg_a, A, neg_b, B, c_mod, C, reuse_a, reuse_b)
  return __builtin_amdgcn_wmma_f32_16x16x32_f16(false, a, false, b, (short)0, c,
                                                false, false);
}

// A-matrix 16x32 f16 fragment: lane holds row (lane&15); K chunks
// [k0+16*hf .. +7] and [k0+16*hf+8 .. +7].
__device__ __forceinline__ v16h a_frag(const _Float16* rowptr, int k0, int hf) {
  v8h lo = *(const v8h*)(rowptr + k0 + 16 * hf);
  v8h hi = *(const v8h*)(rowptr + k0 + 16 * hf + 8);
  v16h a;
#pragma unroll
  for (int i = 0; i < 8; ++i) { a[i] = lo[i]; a[8 + i] = hi[i]; }
  return a;
}

// B-matrix 32x16 fragment built from 16 contiguous f32 (converted to f16):
// lane holds col (lane&15), K = k0 + 16*hf .. +15.
__device__ __forceinline__ v16h b_frag_f32(const float* p) {
  v16h b;
#pragma unroll
  for (int i = 0; i < 16; ++i) b[i] = (_Float16)p[i];
  return b;
}

// ---------------------------------------------------------------------------
// Kernel 0: convert + transpose weights to f16 (B/A fragments contiguous)
// ---------------------------------------------------------------------------
__global__ void wprep_kernel(const float* __restrict__ Wv,
                             const float* __restrict__ Wk,
                             const float* __restrict__ Wq,
                             const float* __restrict__ Wo,
                             _Float16* __restrict__ ws) {
  long idx = (long)blockIdx.x * blockDim.x + threadIdx.x;
  if (idx < HD * HD) {
    int d = (int)(idx / HD), c = (int)(idx % HD);
    ws[OFF_WV + c * HD + d] = (_Float16)Wv[d * HD + c];
    ws[OFF_WK + c * HD + d] = (_Float16)Wk[d * HD + c];
    ws[OFF_WQ + c * HD + d] = (_Float16)Wq[d * HD + c];
  }
  for (long i = idx; i < (long)EMB * EMB; i += (long)gridDim.x * blockDim.x) {
    long e = i / EMB, c = i % EMB;
    ws[OFF_WO + c * EMB + e] = (_Float16)Wo[e * EMB + c];
  }
}

// ---------------------------------------------------------------------------
// Kernel 1: per-head projection, WMMA 16x16x32.  grid=(N*S/64, H), block=128.
// MODE 0 (Q/K): computes (W^T X^T) so the (S,D) row-major store is a packed
//               v8h per column tile.
// MODE 1 (V):   computes (X W) so the transposed (D,S) store is packed.
// ---------------------------------------------------------------------------
template <int MODE>
__global__ __launch_bounds__(128) void proj_kernel(const float* __restrict__ X,
                                                   const _Float16* __restrict__ Wt,
                                                   _Float16* __restrict__ dst) {
  const int lane = threadIdx.x & 31;
  const int wv   = threadIdx.x >> 5;
  const int l16  = lane & 15;
  const int hf   = lane >> 4;
  const int h    = blockIdx.y;

  const long r0 = ((long)blockIdx.x * 4 + wv) * 16;  // flat row over N*S
  const int  n  = (int)(r0 / SEQ);
  const int  s0 = (int)(r0 % SEQ);

  const float* xrow = X + (r0 + l16) * EMB + h * HD;

  if (MODE == 0) {
    // B fragments: columns are the 16 sequence rows; K = head-dim (f32->f16)
    v16h b[2];
#pragma unroll
    for (int ks = 0; ks < 2; ++ks)
      b[ks] = b_frag_f32(xrow + 32 * ks + 16 * hf);

#pragma unroll
    for (int ct = 0; ct < 4; ++ct) {  // 4 x 16 output feature columns
      v8f acc = {};
#pragma unroll
      for (int ks = 0; ks < 2; ++ks) {
        v16h a = a_frag(Wt + (16 * ct + l16) * HD, 32 * ks, hf);
        acc = wmma_f16(a, b[ks], acc);
      }
      v8h st;
#pragma unroll
      for (int e = 0; e < 8; ++e) st[e] = (_Float16)acc[e];
      // row = s0+l16, cols = 16*ct + 8*hf + e (contiguous)
      *(v8h*)(dst + ((long)(n * NH + h) * SEQ + s0 + l16) * HD + 16 * ct + 8 * hf) = st;
    }
  } else {
    // A fragments: rows are sequence rows; K = head-dim (f32->f16)
    v16h a[2];
#pragma unroll
    for (int ks = 0; ks < 2; ++ks) {
      v16h t;
      const float* p = xrow + 32 * ks + 16 * hf;
#pragma unroll
      for (int i = 0; i < 8; ++i) t[i]     = (_Float16)p[i];
#pragma unroll
      for (int i = 0; i < 8; ++i) t[8 + i] = (_Float16)p[8 + i];
      a[ks] = t;
    }
#pragma unroll
    for (int ct = 0; ct < 4; ++ct) {
      v8f acc = {};
#pragma unroll
      for (int ks = 0; ks < 2; ++ks) {
        v16h b = *(const v16h*)(Wt + (16 * ct + l16) * HD + 32 * ks + 16 * hf);
        acc = wmma_f16(a[ks], b, acc);
      }
      v8h st;
#pragma unroll
      for (int e = 0; e < 8; ++e) st[e] = (_Float16)acc[e];
      // V^T: row = feature col (16*ct+l16), cols = s0 + 8*hf + e (contiguous)
      *(v8h*)(dst + ((long)(n * NH + h) * HD + 16 * ct + l16) * SEQ + s0 + 8 * hf) = st;
    }
  }
}

// ---------------------------------------------------------------------------
// Kernel 2: flash attention, transposed orientation.
//   S^T = K * Q^T   (rows = key j, cols = query i -> per-lane softmax stats)
//   O^T = V^T * P^T (rows = head dim d, cols = query i -> packed stores)
// grid=(S/64, H, N), block=128 (4 waves, 16 query rows each).
// ---------------------------------------------------------------------------
__global__ __launch_bounds__(128) void attn_kernel(const float* __restrict__ mask,
                                                   _Float16* ws) {
  __shared__ __align__(32) _Float16 lds_p[4][16 * 32];  // P^T, per-wave private

  const int lane = threadIdx.x & 31;
  const int wv   = threadIdx.x >> 5;
  const int l16  = lane & 15;   // query index i within tile
  const int hf   = lane >> 4;
  const int n    = blockIdx.z;
  const int h    = blockIdx.y;
  const int s0   = blockIdx.x * 64 + wv * 16;

  const _Float16* Q  = ws + OFF_Q + (long)(n * NH + h) * SEQ * HD;
  const _Float16* K  = ws + OFF_K + (long)(n * NH + h) * SEQ * HD;
  const _Float16* Vt = ws + OFF_V + (long)(n * NH + h) * HD * SEQ;
  _Float16* lp = lds_p[wv];

  // Q^T B-fragments (col = query i, K = head dim), resident
  v16h bq[2];
#pragma unroll
  for (int ks = 0; ks < 2; ++ks)
    bq[ks] = *(const v16h*)(Q + (long)(s0 + l16) * HD + 32 * ks + 16 * hf);

  v8f oacc[4];
#pragma unroll
  for (int t = 0; t < 4; ++t) oacc[t] = (v8f){};
  float m = -__builtin_huge_valf();
  float lsum = 0.f;
  const float scale = 0.03125f;  // 1/sqrt(E) = 1/32

  for (int jc = 0; jc < SEQ; jc += 32) {
    if (jc + 32 < SEQ) {
      __builtin_prefetch(K + (long)(jc + 32 + l16) * HD, 0, 0);
      __builtin_prefetch(Vt + (long)l16 * SEQ + jc + 32, 0, 0);
    }
    // S^T tiles: element e -> key j = jc + 16*jt + 8*hf + e, col = query l16
    float x[2][8];
#pragma unroll
    for (int jt = 0; jt < 2; ++jt) {
      v8f acc = {};
#pragma unroll
      for (int ks = 0; ks < 2; ++ks) {
        v16h a = a_frag(K + (long)(jc + 16 * jt + l16) * HD, 32 * ks, hf);
        acc = wmma_f16(a, bq[ks], acc);
      }
      const float* mrow = mask + (long)n * SEQ + jc + 16 * jt + 8 * hf;
#pragma unroll
      for (int e = 0; e < 8; ++e)
        x[jt][e] = (mrow[e] == 0.f) ? -3.125e18f : acc[e] * scale;
    }
    // per-query-row stats: local reduce over 16 elems + one cross-half shuffle
    float lmax = x[0][0];
#pragma unroll
    for (int jt = 0; jt < 2; ++jt)
#pragma unroll
      for (int e = 0; e < 8; ++e) lmax = fmaxf(lmax, x[jt][e]);
    lmax = fmaxf(lmax, __shfl_xor(lmax, 16, 32));

    const float mn   = fmaxf(m, lmax);
    const float corr = __expf(m - mn);
    m = mn;

    float p[2][8], ls = 0.f;
#pragma unroll
    for (int jt = 0; jt < 2; ++jt)
#pragma unroll
      for (int e = 0; e < 8; ++e) {
        const float pe = __expf(x[jt][e] - mn);
        p[jt][e] = pe;
        ls += pe;
      }
    ls += __shfl_xor(ls, 16, 32);
    lsum = lsum * corr + ls;
#pragma unroll
    for (int t = 0; t < 4; ++t)
#pragma unroll
      for (int e = 0; e < 8; ++e) oacc[t][e] *= corr;

    // P^T -> LDS, column-major per query row: lp[i*32 + j_local], packed v8h
#pragma unroll
    for (int jt = 0; jt < 2; ++jt) {
      v8h ph;
#pragma unroll
      for (int e = 0; e < 8; ++e) ph[e] = (_Float16)p[jt][e];
      *(v8h*)(lp + l16 * 32 + 16 * jt + 8 * hf) = ph;
    }
    // P^T B-fragment: col = query l16, K = j_local 16*hf..+15 (contiguous)
    v16h bp = *(const v16h*)(lp + l16 * 32 + 16 * hf);

    // O^T += V^T * P^T : A rows = head-dim d, K = j (contiguous in Vt rows)
#pragma unroll
    for (int t = 0; t < 4; ++t) {
      v16h a = a_frag(Vt + (long)(16 * t + l16) * SEQ, jc, hf);
      oacc[t] = wmma_f16(a, bp, oacc[t]);
    }
  }

  // normalize (per-lane scalar) and emit packed v8h rows of O (N,S,E)
  const float inv = 1.0f / lsum;
#pragma unroll
  for (int t = 0; t < 4; ++t) {
    v8h oh;
#pragma unroll
    for (int e = 0; e < 8; ++e) oh[e] = (_Float16)(oacc[t][e] * inv);
    // row = s0+l16, cols = h*HD + 16*t + 8*hf + e (contiguous)
    *(v8h*)(ws + OFF_O + ((long)n * SEQ + s0 + l16) * EMB + h * HD + 16 * t + 8 * hf) = oh;
  }
}

// ---------------------------------------------------------------------------
// Kernel 3: out^T = Wo^T * O^T  (+ bo), packed f32 stores.
// grid=(E/32, N*S/64), block=128.  Wave: 16 seq rows x 32 out cols.
// ---------------------------------------------------------------------------
__global__ __launch_bounds__(128) void outproj_kernel(const _Float16* __restrict__ ws,
                                                      const float* __restrict__ bo,
                                                      float* __restrict__ out) {
  const int lane = threadIdx.x & 31;
  const int wv   = threadIdx.x >> 5;
  const int l16  = lane & 15;   // seq row within tile
  const int hf   = lane >> 4;
  const int c0   = blockIdx.x * 32;
  const long r0  = ((long)blockIdx.y * 4 + wv) * 16;

  const _Float16* O   = ws + OFF_O;
  const _Float16* Wot = ws + OFF_WO;

  v8f acc[2] = {(v8f){}, (v8f){}};
  for (int e0 = 0; e0 < EMB; e0 += 32) {
    // B fragment: col = seq row, K = embed dim (contiguous in O rows)
    v16h b = *(const v16h*)(O + (r0 + l16) * EMB + e0 + 16 * hf);
#pragma unroll
    for (int ct = 0; ct < 2; ++ct) {
      v16h a = a_frag(Wot + (long)(c0 + 16 * ct + l16) * EMB, e0, hf);
      acc[ct] = wmma_f16(a, b, acc[ct]);
    }
  }
#pragma unroll
  for (int ct = 0; ct < 2; ++ct) {
    const int cbase = c0 + 16 * ct + 8 * hf;  // element e -> col cbase+e
    float4 r0v, r1v;
    const float4 b0 = *(const float4*)(bo + cbase);
    const float4 b1 = *(const float4*)(bo + cbase + 4);
    r0v.x = acc[ct][0] + b0.x; r0v.y = acc[ct][1] + b0.y;
    r0v.z = acc[ct][2] + b0.z; r0v.w = acc[ct][3] + b0.w;
    r1v.x = acc[ct][4] + b1.x; r1v.y = acc[ct][5] + b1.y;
    r1v.z = acc[ct][6] + b1.z; r1v.w = acc[ct][7] + b1.w;
    float* op = out + (r0 + l16) * EMB + cbase;
    *(float4*)op       = r0v;
    *(float4*)(op + 4) = r1v;
  }
}

// ---------------------------------------------------------------------------
extern "C" void kernel_launch(void* const* d_in, const int* in_sizes, int n_in,
                              void* d_out, int out_size, void* d_ws, size_t ws_size,
                              hipStream_t stream) {
  const float* values = (const float*)d_in[0];
  const float* keys   = (const float*)d_in[1];
  const float* query  = (const float*)d_in[2];
  const float* mask   = (const float*)d_in[3];
  const float* Wv     = (const float*)d_in[4];
  const float* Wk     = (const float*)d_in[5];
  const float* Wq     = (const float*)d_in[6];
  const float* Wo     = (const float*)d_in[7];
  const float* bo     = (const float*)d_in[8];
  float* out          = (float*)d_out;
  _Float16* ws        = (_Float16*)d_ws;

  wprep_kernel<<<1024, 256, 0, stream>>>(Wv, Wk, Wq, Wo, ws);

  dim3 gproj(NB * SEQ / 64, NH);
  proj_kernel<0><<<gproj, 128, 0, stream>>>(query,  ws + OFF_WQ, ws + OFF_Q);
  proj_kernel<0><<<gproj, 128, 0, stream>>>(keys,   ws + OFF_WK, ws + OFF_K);
  proj_kernel<1><<<gproj, 128, 0, stream>>>(values, ws + OFF_WV, ws + OFF_V);

  dim3 gattn(SEQ / 64, NH, NB);
  attn_kernel<<<gattn, 128, 0, stream>>>(mask, ws);

  dim3 gout(EMB / 32, NB * SEQ / 64);
  outproj_kernel<<<gout, 128, 0, stream>>>(ws, bo, out);
}
